// PicanetL_36876589204043
// MI455X (gfx1250) — matrix operand order repaired and checked by reading, not compiled, not run on hardware
//
#include <hip/hip_runtime.h>

typedef __attribute__((ext_vector_type(16))) __bf16 v16bf;
typedef __attribute__((ext_vector_type(8)))  __bf16 v8bf;
typedef __attribute__((ext_vector_type(8)))  float  v8f;
typedef unsigned int u32x4 __attribute__((ext_vector_type(4)));
typedef unsigned int u32x4a __attribute__((ext_vector_type(4), aligned(4)));

// int4 pointers in explicit address spaces for the async-to-LDS builtin:
// signature (probe-confirmed via diagnostic): (v4i as1*, v4i as3*, imm, imm)
typedef __attribute__((ext_vector_type(4))) int v4i;
typedef __attribute__((address_space(1))) v4i as1v4i;
typedef __attribute__((address_space(3))) v4i as3v4i;

#define BB   8
#define CC   64
#define HH   112
#define WW   112
#define MID  256
#define NTAP 25
#define KTOT 1600        // CC * NTAP (conv1 reduction)
#define KC_N 50          // KTOT / 32
#define KPAD 1608        // padded K row (bf16) for LDS bank stagger
#define NPX  16          // pixels per GEMM N-tile
#define NT   2           // row-tiles per workgroup (N reuse of A fragments)
#define MPAD 264         // padded feature row (bf16) for conv2 B slab
#define XPH  120         // padded image height (4 halo each side)
#define XPW  120         // padded image width

__device__ __forceinline__ v16bf bjoin(v8bf lo, v8bf hi) {
  return __builtin_shufflevector(lo, hi, 0, 1, 2, 3, 4, 5, 6, 7,
                                 8, 9, 10, 11, 12, 13, 14, 15);
}
__device__ __forceinline__ v8f wmma_bf16(v16bf a, v16bf b, v8f c) {
  return __builtin_amdgcn_wmma_f32_16x16x32_bf16(false, a, false, b,
                                                 (short)0, c, false, false);
}

// ---------------------------------------------------------------------------
// prep: conv1 weights fp32 [256,64,5,5] -> bf16 WMMA A-fragments.
// frag = mtile*50 + kc ; element = (frag*32 + lane)*16 + t ; kk = tap*64 + c
// A k order used here (contiguous, same as B): lane holds M=lane%16; lo lanes K={0..15},
// hi lanes K={16..31} of the 32-wide chunk.
// ---------------------------------------------------------------------------
__global__ __launch_bounds__(256) void picanet_prep_w1(
    const float* __restrict__ w1, __bf16* __restrict__ w1b) {
  int idx = blockIdx.x * 256 + threadIdx.x;
  if (idx >= MID * KTOT) return;
  int t    = idx & 15;
  int lane = (idx >> 4) & 31;
  int frag = idx >> 9;
  int kc   = frag % KC_N;
  int mt   = frag / KC_N;
  int hsel = lane >> 4;
  int m    = mt * 16 + (lane & 15);
  int koff = t + hsel * 16;
  int kk   = kc * 32 + koff;
  int tap  = kk >> 6;
  int c    = kk & 63;
  w1b[idx] = (__bf16)w1[(m * CC + c) * NTAP + tap];
}

// ---------------------------------------------------------------------------
// prep: conv2 weights fp32 [25,256] -> bf16 A-fragments, M padded to 32.
// frag = mt2*8 + kc2 ; 8192 elements total.
// ---------------------------------------------------------------------------
__global__ __launch_bounds__(256) void picanet_prep_w2(
    const float* __restrict__ w2, __bf16* __restrict__ w2f) {
  int idx = blockIdx.x * 256 + threadIdx.x;
  if (idx >= 2 * 8 * 32 * 16) return;
  int t    = idx & 15;
  int lane = (idx >> 4) & 31;
  int frag = idx >> 9;
  int k2   = frag & 7;
  int mt2  = frag >> 3;
  int hsel = lane >> 4;
  int m    = mt2 * 16 + (lane & 15);
  int koff = t + hsel * 16;
  int kk   = k2 * 32 + koff;
  float v  = (m < NTAP) ? w2[m * MID + kk] : 0.0f;
  w2f[idx] = (__bf16)v;
}

// prep: fused logit bias  lbias[o] = b2[o] + sum_m w2[o,m]*b1[m]
__global__ void picanet_prep_lbias(const float* __restrict__ w2,
                                   const float* __restrict__ b1,
                                   const float* __restrict__ b2,
                                   float* __restrict__ lbias) {
  int o = threadIdx.x;
  if (o >= NTAP) return;
  float s = b2[o];
  for (int m = 0; m < MID; ++m) s = fmaf(w2[o * MID + m], b1[m], s);
  lbias[o] = s;
}

// prep: x fp32 [8,64,112,112] -> zero-padded bf16 [8,64,120,120] (+4 halo)
__global__ __launch_bounds__(256) void picanet_prep_x(
    const float* __restrict__ x, __bf16* __restrict__ xpad) {
  int idx = blockIdx.x * 256 + threadIdx.x;
  if (idx >= BB * CC * XPH * XPW) return;
  int xx = idx % XPW;
  int y  = (idx / XPW) % XPH;
  int bc = idx / (XPW * XPH);
  int sx = xx - 4, sy = y - 4;
  float v = 0.0f;
  if ((unsigned)sy < (unsigned)HH && (unsigned)sx < (unsigned)WW)
    v = x[(size_t)bc * (HH * WW) + sy * WW + sx];
  xpad[idx] = (__bf16)v;
}

// ---------------------------------------------------------------------------
// Fused main: conv1 (bf16 WMMA, 2 row-tiles share A) + conv2 (bf16 WMMA)
// + softmax(25) + 25-tap weighted sum. One WG per (b, row-pair, 16-px tile).
// ---------------------------------------------------------------------------
__global__ __launch_bounds__(256) void picanet_main(
    const __bf16* __restrict__ xpad, const __bf16* __restrict__ w1b,
    const __bf16* __restrict__ w2f,  const float* __restrict__ lbias,
    float* __restrict__ out) {
  // 102912 B LDS, reused across phases:
  //  phase1: im2col patch  ldsB[NT][NPX][KPAD] bf16
  //  phase2: featbf[NT][NPX][MPAD] bf16 (16896 B) | attn (3200 B) | ldsW (16 KB)
  __shared__ __align__(16) unsigned char smem[NT * NPX * KPAD * 2];
  __bf16*         ldsB   = (__bf16*)smem;
  unsigned short* ldsBu  = (unsigned short*)smem;
  __bf16*         featbf = (__bf16*)smem;
  float*          attn   = (float*)(smem + NT * NPX * MPAD * 2);   // 16896
  __bf16*         ldsW   = (__bf16*)(smem + 20480);                 // 16 KB

  const int tid  = threadIdx.x;
  const int lane = tid & 31;
  const int wave = tid >> 5;

  int bid = blockIdx.x;
  int b   = bid / ((HH / NT) * (WW / NPX));
  int rem = bid % ((HH / NT) * (WW / NPX));
  int h0  = (rem / (WW / NPX)) * NT;
  int w0  = (rem % (WW / NPX)) * NPX;

  // ---- phase 1a: stage im2col patch (bf16, padded -> no bounds checks) ---
  // One iteration: 8 pixels of one kk row for BOTH row-tiles (tile1 is the
  // same source address +1 image row), align-4 b128 loads + b16 LDS scatter.
  const __bf16* xb = xpad + (size_t)b * (CC * XPH * XPW);
  for (int s = tid; s < KTOT * 2; s += 256) {
    int kk  = s >> 1;
    int phx = (s & 1) << 3;
    int tap = kk >> 6;
    int c   = kk & 63;
    int i = tap / 5, j = tap - i * 5;
    int row = h0 + 2 * i;                  // padded y  = (h+2i-4)+4
    int col = w0 + 2 * j + phx;            // padded x0 = (w0+2j-4)+4
    const __bf16* p = xb + (c * XPH + row) * XPW + col;
    u32x4a s0 = *(const u32x4a*)p;
    u32x4a s1 = *(const u32x4a*)(p + XPW); // row-tile 1 = next image row
#pragma unroll
    for (int q = 0; q < 4; ++q) {
      unsigned int d0 = s0[q];
      unsigned int d1 = s1[q];
      int px = phx + 2 * q;
      ldsBu[px * KPAD + kk]               = (unsigned short)d0;
      ldsBu[(px + 1) * KPAD + kk]         = (unsigned short)(d0 >> 16);
      ldsBu[(NPX + px) * KPAD + kk]       = (unsigned short)d1;
      ldsBu[(NPX + px + 1) * KPAD + kk]   = (unsigned short)(d1 >> 16);
    }
  }
  __syncthreads();

  // ---- phase 1b: conv1 implicit GEMM, each A fragment feeds 2 N-tiles ----
  const int mt0  = wave;
  const int mt1  = wave + 8;
  const int col  = lane & 15;
  const int hsel = lane >> 4;
  // byte-stepped fragment pointers: +1 fragment (= 512 bf16) per k-chunk
  const __bf16* pa0 = w1b + ((size_t)(mt0 * KC_N) * 32 + lane) * 16;
  const __bf16* pa1 = w1b + ((size_t)(mt1 * KC_N) * 32 + lane) * 16;
  const __bf16* pb0 = ldsB + col * KPAD + hsel * 16;
  const __bf16* pb1 = pb0 + NPX * KPAD;

  v8f acc00 = {}, acc01 = {}, acc10 = {}, acc11 = {};
  for (int kc = 0; kc < KC_N; ++kc) {
    v16bf b0 = bjoin(((const v8bf*)pb0)[0], ((const v8bf*)pb0)[1]);
    v16bf b1 = bjoin(((const v8bf*)pb1)[0], ((const v8bf*)pb1)[1]);
    v16bf a0 = *(const v16bf*)pa0;
    v16bf a1 = *(const v16bf*)pa1;
    __builtin_prefetch(pa0 + 512, 0, 3);   // next k-chunk (speculative ok)
    pa0 += 512;
    pa1 += 512;
    pb0 += 32;
    pb1 += 32;
    // ordering keeps one shared operand between consecutive WMMAs
    acc00 = wmma_bf16(a0, b0, acc00);
    acc01 = wmma_bf16(a0, b1, acc01);
    acc11 = wmma_bf16(a1, b1, acc11);
    acc10 = wmma_bf16(a1, b0, acc10);
  }
  __syncthreads();   // ldsB dead; LDS reused below

  // ---- phase 2a: spill features as bf16 conv2-B slab [tile][px][MPAD] ----
  // C/D layout: VGPR r -> M = mt*16 + hsel*8 + r, N = col.
  {
    int mb0 = mt0 * 16 + hsel * 8;
    int mb1 = mt1 * 16 + hsel * 8;
    v8bf f00, f01, f10, f11;
#pragma unroll
    for (int r = 0; r < 8; ++r) {
      f00[r] = (__bf16)acc00[r];
      f01[r] = (__bf16)acc01[r];
      f10[r] = (__bf16)acc10[r];
      f11[r] = (__bf16)acc11[r];
    }
    *(v8bf*)&featbf[(0 * NPX + col) * MPAD + mb0] = f00;
    *(v8bf*)&featbf[(1 * NPX + col) * MPAD + mb0] = f01;
    *(v8bf*)&featbf[(0 * NPX + col) * MPAD + mb1] = f10;
    *(v8bf*)&featbf[(1 * NPX + col) * MPAD + mb1] = f11;
  }

  // ---- phase 2b: async-copy the 16 KB w2 fragment table L2 -> LDS --------
  {
    char* src = (char*)w2f;       // builtin takes non-const pointers
    char* dst = (char*)ldsW;
#if __has_builtin(__builtin_amdgcn_global_load_async_to_lds_b128)
    for (int off = tid * 16; off < 16384; off += 256 * 16)
      __builtin_amdgcn_global_load_async_to_lds_b128(
          (as1v4i*)(src + off), (as3v4i*)(dst + off), 0, 0);
#if __has_builtin(__builtin_amdgcn_s_wait_asynccnt)
    __builtin_amdgcn_s_wait_asynccnt(0);
#else
    asm volatile("s_wait_asynccnt 0" ::: "memory");
#endif
#else
    for (int off = tid * 16; off < 16384; off += 256 * 16)
      *(u32x4*)(dst + off) = *(const u32x4*)(src + off);
#endif
  }
  __syncthreads();

  // ---- phase 2c: conv2 as WMMA (K=256, M=32 padded, N=16) ----------------
  // waves 0-3 compute tile0, waves 4-7 tile1 (wave-uniform addressing).
  {
    int tsel = wave >> 2;
    const __bf16* fb = featbf + (tsel * NPX + col) * MPAD + hsel * 16;
    v8f l0 = {}, l1 = {};
#pragma unroll
    for (int k2 = 0; k2 < 8; ++k2) {
      const v8bf* bp  = (const v8bf*)(fb + k2 * 32);
      v16bf bf_ = bjoin(bp[0], bp[1]);
      const v8bf* a0p = (const v8bf*)(ldsW + ((0 + k2) * 32 + lane) * 16);
      const v8bf* a1p = (const v8bf*)(ldsW + ((8 + k2) * 32 + lane) * 16);
      v16bf a0 = bjoin(a0p[0], a0p[1]);
      v16bf a1 = bjoin(a1p[0], a1p[1]);
      l0 = wmma_bf16(a0, bf_, l0);
      l1 = wmma_bf16(a1, bf_, l1);
    }
    if ((wave & 3) == 0) {       // wave 0 -> tile0 logits, wave 4 -> tile1
      float* at = attn + tsel * (NTAP * NPX);
#pragma unroll
      for (int r = 0; r < 8; ++r) {
        int o0 = hsel * 8 + r;
        at[o0 * NPX + col] = l0[r];
        int o1 = 16 + hsel * 8 + r;
        if (o1 < NTAP) at[o1 * NPX + col] = l1[r];
      }
    }
  }
  __syncthreads();

  // ---- phase 2d: softmax over 25 taps (+fused logit bias), per pixel -----
  if (tid < NT * NPX) {
    int tsel = tid >> 4, px = tid & 15;
    float* at = attn + tsel * (NTAP * NPX);
    float mx = -3.4e38f;
    for (int o = 0; o < NTAP; ++o) {
      float v = at[o * NPX + px] + lbias[o];
      at[o * NPX + px] = v;
      mx = fmaxf(mx, v);
    }
    float ssum = 0.0f;
    for (int o = 0; o < NTAP; ++o) {
      float e = __expf(at[o * NPX + px] - mx);
      at[o * NPX + px] = e;
      ssum += e;
    }
    float inv = 1.0f / ssum;
    for (int o = 0; o < NTAP; ++o) at[o * NPX + px] *= inv;
  }
  __syncthreads();

  // ---- phase 3: 25-tap weighted neighborhood sum (padded bf16 source) ----
  for (int t = tid; t < NT * CC * NPX; t += 256) {
    int tile = t >> 10;
    int c    = (t >> 4) & 63;
    int px   = t & 15;
    int h    = h0 + tile;
    const __bf16* xc = xpad + ((size_t)b * CC + c) * (XPH * XPW);
    const float*  at = attn + tile * (NTAP * NPX);
    float acc = 0.0f;
#pragma unroll
    for (int tap = 0; tap < NTAP; ++tap) {
      int i = tap / 5, j = tap - i * 5;
      float v = (float)xc[(h + 2 * i) * XPW + (w0 + px + 2 * j)];
      acc = fmaf(v, at[tap * NPX + px], acc);
    }
    out[(((size_t)b * CC + c) * HH + h) * WW + w0 + px] = acc;
  }
}

// ---------------------------------------------------------------------------
extern "C" void kernel_launch(void* const* d_in, const int* in_sizes, int n_in,
                              void* d_out, int out_size, void* d_ws,
                              size_t ws_size, hipStream_t stream) {
  const float* x  = (const float*)d_in[0];   // [8,64,112,112]
  const float* w1 = (const float*)d_in[1];   // [256,64,5,5]
  const float* b1 = (const float*)d_in[2];   // [256]
  const float* w2 = (const float*)d_in[3];   // [25,256,1,1]
  const float* b2 = (const float*)d_in[4];   // [25]
  float* out = (float*)d_out;

  // workspace carve-up (bytes)
  char* ws = (char*)d_ws;
  __bf16* w1b   = (__bf16*)(ws + 0);         //   819200 B
  __bf16* w2f   = (__bf16*)(ws + 819200);    //    16384 B
  float*  lbias = (float*) (ws + 835584);    //      100 B
  __bf16* xpad  = (__bf16*)(ws + 835840);    // 14745600 B  (total ~15.6 MB)

  picanet_prep_w1<<<(MID * KTOT + 255) / 256, 256, 0, stream>>>(w1, w1b);
  picanet_prep_w2<<<(2 * 8 * 32 * 16 + 255) / 256, 256, 0, stream>>>(w2, w2f);
  picanet_prep_lbias<<<1, 32, 0, stream>>>(w2, b1, b2, lbias);
  picanet_prep_x<<<(BB * CC * XPH * XPW + 255) / 256, 256, 0, stream>>>(x, xpad);

  int grid = BB * (HH / NT) * (WW / NPX);    // 8*56*7 = 3136
  picanet_main<<<grid, 256, 0, stream>>>(xpad, w1b, w2f, lbias, out);
}
